// TrafficNodeModel_1657857376695
// MI455X (gfx1250) — matrix operand order repaired and checked by reading, not compiled, not run on hardware
//
#include <hip/hip_runtime.h>
#include <hip/hip_bf16.h>

// ---------------- types ----------------
typedef unsigned int u32;
typedef _Float16 h16;
typedef __attribute__((ext_vector_type(16))) _Float16 v16h;
typedef __attribute__((ext_vector_type(8)))  float    v8f;
typedef __attribute__((ext_vector_type(4)))  u32      u32x4;

#define BB   128
#define SS   2048
#define KK   64
#define EE   128
#define HH   256
#define SP   (SS + 6)   // padded sequence rows (3 halo each side)

union FragH { v16h v; h16 h[16]; u32 u[8]; u32x4 q[2]; };

#define WMMA(a, b, c) __builtin_amdgcn_wmma_f32_16x16x32_f16(false, (a), false, (b), (short)0, (c), false, false)

// A fragment (16x32 f16) from row-major f16: two contiguous 16B runs -> 2x ds/global b128.
// lane L: M = m0 + L%16 ; K = k0 + 16*(L/16) + {0..15}
__device__ __forceinline__ v16h load_fragA(const h16* base, int lda, int m0, int k0) {
  int lane = threadIdx.x & 31;
  int m  = m0 + (lane & 15);
  int kb = k0 + ((lane >> 4) << 4);
  const h16* row = base + m * lda;
  FragH f;
  f.q[0] = *(const u32x4*)(row + kb);
  f.q[1] = *(const u32x4*)(row + kb + 8);
  return f.v;
}

// B fragment (32x16 f16) from fragment-major global layout:
// halves stored at [((k/32)*N + n)*16 + (k%32)/2]*2 + (k&1), so each lane's 8
// dwords for one 32-K chunk are one contiguous 32B run -> 2x global_load_b128.
// lane L: N = n0 + L%16 ; K = k0 + 16*(L/16) + {0..15}
__device__ __forceinline__ v16h load_fragB(const h16* bp, int N, int k0, int n0) {
  int lane = threadIdx.x & 31;
  int n  = n0 + (lane & 15);
  int j0 = (lane >> 4) << 3;                  // dword offset 0 or 8 within chunk
  const h16* base = bp + ((size_t)((k0 >> 5) * N + n) * 16 + j0) * 2;
  FragH f;
  f.q[0] = *(const u32x4*)(base);
  f.q[1] = *(const u32x4*)(base + 8);
  return f.v;
}

// ---------------- k0: weight prep (f16 fragment-major), zero accumulators/halos ----------------
__global__ __launch_bounds__(256) void k0_prep(
    const float* __restrict__ kernel_emb, const float* __restrict__ proj_w,
    const float* __restrict__ c3w, const float* __restrict__ c3b,
    const float* __restrict__ c5w, const float* __restrict__ c5b,
    const float* __restrict__ c7w, const float* __restrict__ c7b,
    h16* keQ, h16* projTQ, h16* W7Q, float* hbias,
    float* bnsum, float* bnsumsq, h16* seqP)
{
  int tid = blockIdx.x * blockDim.x + threadIdx.x;
  int nth = gridDim.x * blockDim.x;
  // kernel_emb [K=64][E=128] -> B (N=128)
  for (int i = tid; i < KK * EE; i += nth) {
    int k = i >> 7, n = i & 127;
    keQ[(((size_t)(k >> 5) * EE + n) * 16 + ((k & 31) >> 1)) * 2 + (k & 1)] = (h16)kernel_emb[i];
  }
  // proj^T: B[k=e_in][n=e_out] = proj_w[n][e_in]  (N=128)
  for (int i = tid; i < EE * EE; i += nth) {
    int k = i >> 7, n = i & 127;
    projTQ[(((size_t)(k >> 5) * EE + n) * 16 + ((k & 31) >> 1)) * 2 + (k & 1)] = (h16)proj_w[n * EE + k];
  }
  // merged 7-tap conv weights: W7[dk][k=e][n=h], zero-padded taps (N=256)
  for (int i = tid; i < 7 * EE * HH; i += nth) {
    int dk = i / (EE * HH);
    int rem = i - dk * (EE * HH);
    int k = rem >> 8, n = rem & 255;
    float v = 0.f;
    if (n < 85)       { int d = dk - 2; if (d >= 0 && d < 3) v = c3w[(n * EE + k) * 3 + d]; }
    else if (n < 170) { int d = dk - 1; if (d >= 0 && d < 5) v = c5w[((n - 85) * EE + k) * 5 + d]; }
    else              {                                      v = c7w[((n - 170) * EE + k) * 7 + dk]; }
    W7Q[(size_t)dk * (EE * HH) + (((size_t)(k >> 5) * HH + n) * 16 + ((k & 31) >> 1)) * 2 + (k & 1)] = (h16)v;
  }
  for (int i = tid; i < HH; i += nth) {
    hbias[i] = (i < 85) ? c3b[i] : (i < 170) ? c5b[i - 85] : c7b[i - 170];
    bnsum[i] = 0.f; bnsumsq[i] = 0.f;
  }
  // zero conv halos of seqP
  for (int i = tid; i < BB * 6 * EE; i += nth) {
    int b = i / (6 * EE);
    int r = (i / EE) % 6;
    int e = i & 127;
    int row = (r < 3) ? r : (SS + 3) + (r - 3);
    seqP[((size_t)b * SP + row) * EE + e] = (h16)0.f;
  }
}

// ---------------- k1: gaussian embed + sign + projection ----------------
__global__ __launch_bounds__(256) void k1_embed_proj(
    const float* __restrict__ raw_seq, const float* __restrict__ mu,
    const float* __restrict__ log_sigma, const float* __restrict__ sign_emb,
    const float* __restrict__ proj_b,
    const h16* __restrict__ keQ, const h16* __restrict__ projTQ,
    h16* __restrict__ seqP)
{
  __shared__ __align__(16) h16 A1[128 * KK];   // 16 KB normalized weights (f16)
  __shared__ __align__(16) h16 T[128 * EE];    // 32 KB mag+sign embedding (f16)
  __shared__ float muS[KK], i2s2[KK];
  __shared__ float signS[3 * EE];
  __shared__ int   sidx[128];
  __shared__ float pbS[EE];

  int t = threadIdx.x;
  int r0 = blockIdx.x * 128;            // 128 tokens per block, within one batch
  int b  = r0 / SS;
  int s0 = r0 % SS;

  if (t < KK) {
    float ls = log_sigma[t];
    float sg = log1pf(__expf(ls)) + 1.0f;   // softplus + MIN_SIGMA
    i2s2[t] = 1.0f / (2.0f * sg * sg);
    muS[t] = mu[t];
  }
  for (int i = t; i < 3 * EE; i += 256) signS[i] = sign_emb[i];
  if (t < EE) pbS[t] = proj_b[t];
  __syncthreads();

  if (t < 128) {
    float x  = raw_seq[r0 + t];
    float xa = fabsf(x);
    sidx[t] = (x < 0.f) ? 1 : ((x > 0.f) ? 2 : 0);
    float sum = 1e-8f;
#pragma unroll 8
    for (int k = 0; k < KK; ++k) {
      float d = xa - muS[k];
      sum += __expf(-d * d * i2s2[k]);
    }
    float rs = 1.0f / sum;
#pragma unroll 8
    for (int k = 0; k < KK; ++k) {
      float d = xa - muS[k];
      A1[t * KK + k] = (h16)(__expf(-d * d * i2s2[k]) * rs);
    }
  }
  __syncthreads();

  int wave = t >> 5, lane = t & 31;
  int m0 = wave * 16;
  int nloc = lane & 15;
  int mh = (lane >> 4) * 8;

  // GEMM1: [128x64] @ [64x128] + sign_emb -> T (f16); A reused across all 8 N-tiles
  {
    v8f acc[8] = {};
#pragma unroll
    for (int kc = 0; kc < KK; kc += 32) {
      v16h a = load_fragA(A1, KK, m0, kc);
#pragma unroll
      for (int nt = 0; nt < 8; ++nt) {
        v16h bm = load_fragB(keQ, EE, kc, nt * 16);
        acc[nt] = WMMA(a, bm, acc[nt]);
      }
    }
#pragma unroll
    for (int nt = 0; nt < 8; ++nt) {
      int n = nt * 16 + nloc;
#pragma unroll
      for (int j = 0; j < 8; ++j) {
        int m = m0 + mh + j;
        T[m * EE + n] = (h16)(acc[nt][j] + signS[sidx[m] * EE + n]);
      }
    }
  }
  __syncthreads();

  // GEMM2: T @ proj^T + proj_b -> seqP (padded, f16)
  {
    v8f acc[8] = {};
#pragma unroll
    for (int kc = 0; kc < EE; kc += 32) {
      v16h a = load_fragA(T, EE, m0, kc);
#pragma unroll
      for (int nt = 0; nt < 8; ++nt) {
        v16h bm = load_fragB(projTQ, EE, kc, nt * 16);
        acc[nt] = WMMA(a, bm, acc[nt]);
      }
    }
    size_t rowbase = ((size_t)b * SP + (s0 + 3)) * EE;
#pragma unroll
    for (int nt = 0; nt < 8; ++nt) {
      int n = nt * 16 + nloc;
#pragma unroll
      for (int j = 0; j < 8; ++j) {
        int m = m0 + mh + j;
        seqP[rowbase + (size_t)m * EE + n] = (h16)(acc[nt][j] + pbS[n]);
      }
    }
  }
}

// ---------------- k2: merged 7-tap conv as shifted GEMMs + ReLU + BN stats ----------------
// 128 threads = 4 waves; wave owns 2 M-strips (32 rows); N processed in 2 halves of 128.
// Inner loop: per (tap, k-chunk): 2 A frags (4 ds_b128), 8 B frags (16 glb_b128), 16 WMMA.
__global__ __launch_bounds__(128) void k2_conv(
    const h16* __restrict__ seqP, const h16* __restrict__ W7Q,
    const float* __restrict__ hbias,
    h16* __restrict__ hbuf, float* bnsum, float* bnsumsq)
{
  __shared__ __align__(16) h16 Ap[134 * EE];   // 33.5 KB: 128 rows + 6 halo rows
  __shared__ float bnl[2 * HH];

  int t = threadIdx.x;
  int r0 = blockIdx.x * 128;
  int b  = r0 / SS;
  int s0 = r0 % SS;

  // stage padded rows s0..s0+133 into LDS
  const u32* src = (const u32*)(seqP + ((size_t)b * SP + s0) * EE);
  u32* dst = (u32*)Ap;
  for (int i = t; i < 134 * (EE / 2); i += 128) dst[i] = src[i];
  for (int i = t; i < 2 * HH; i += 128) bnl[i] = 0.f;
  __syncthreads();

  int wave = t >> 5, lane = t & 31;
  int m0 = wave * 32;          // two 16-row strips: m0 and m0+16
  int nloc = lane & 15;
  int mh = (lane >> 4) * 8;

  for (int half = 0; half < 2; ++half) {
    v8f acc0[8] = {};
    v8f acc1[8] = {};
    for (int dk = 0; dk < 7; ++dk) {
      const h16* w = W7Q + (size_t)dk * (EE * HH);
#pragma unroll
      for (int kc = 0; kc < EE; kc += 32) {
        v16h a0 = load_fragA(Ap, EE, m0 + dk,      kc);   // shifted rows = conv taps
        v16h a1 = load_fragA(Ap, EE, m0 + 16 + dk, kc);
#pragma unroll
        for (int nt = 0; nt < 8; ++nt) {
          v16h bm = load_fragB(w, HH, kc, half * 128 + nt * 16);
          acc0[nt] = WMMA(a0, bm, acc0[nt]);
          acc1[nt] = WMMA(a1, bm, acc1[nt]);
        }
      }
    }
    // epilogue: bias + ReLU + f16 store + BN partial sums
#pragma unroll
    for (int nt = 0; nt < 8; ++nt) {
      int n = half * 128 + nt * 16 + nloc;
      float bia = hbias[n];
      float lsum = 0.f, lsq = 0.f;
#pragma unroll
      for (int j = 0; j < 8; ++j) {
        int m = m0 + mh + j;
        float v = fmaxf(acc0[nt][j] + bia, 0.f);
        hbuf[((size_t)b * SS + s0 + m) * HH + n] = (h16)v;
        lsum += v; lsq += v * v;
        int m2 = m + 16;
        float v2 = fmaxf(acc1[nt][j] + bia, 0.f);
        hbuf[((size_t)b * SS + s0 + m2) * HH + n] = (h16)v2;
        lsum += v2; lsq += v2 * v2;
      }
      atomicAdd(&bnl[n], lsum);
      atomicAdd(&bnl[HH + n], lsq);
    }
  }
  __syncthreads();
  for (int i = t; i < HH; i += 128) {
    atomicAdd(&bnsum[i],   bnl[i]);
    atomicAdd(&bnsumsq[i], bnl[HH + i]);
  }
}

// ---------------- k3: FiLM MLP + fold BN into per-(b,h) affine ----------------
__global__ __launch_bounds__(256) void k3_film(
    const float* __restrict__ raw_stats,
    const float* __restrict__ sw1, const float* __restrict__ sb1,
    const float* __restrict__ sw2, const float* __restrict__ sb2,
    const float* __restrict__ bn_gamma, const float* __restrict__ bn_beta,
    const float* __restrict__ bnsum, const float* __restrict__ bnsumsq,
    float* __restrict__ Acoef, float* __restrict__ Ccoef)
{
  __shared__ float hid[64];
  __shared__ float film[2 * HH];
  int b = blockIdx.x, t = threadIdx.x;
  if (t < 64) {
    float a = sb1[t];
#pragma unroll
    for (int k = 0; k < 5; ++k) a += raw_stats[b * 5 + k] * sw1[t * 5 + k];
    hid[t] = fmaxf(a, 0.f);
  }
  __syncthreads();
  for (int o = t; o < 2 * HH; o += 256) {
    float a = sb2[o];
#pragma unroll 8
    for (int k = 0; k < 64; ++k) a += hid[k] * sw2[o * 64 + k];
    film[o] = a;
  }
  __syncthreads();
  if (t < HH) {
    const float inv_n = 1.0f / ((float)BB * (float)SS);
    float mean = bnsum[t] * inv_n;
    float var  = bnsumsq[t] * inv_n - mean * mean;
    float istd = rsqrtf(var + 1e-5f);
    float g = bn_gamma[t] * istd;
    float s = 1.0f + film[t];           // (1 + gamma_film)
    Acoef[b * HH + t] = g * s;
    Ccoef[b * HH + t] = (bn_beta[t] - mean * g) * s + film[HH + t];
  }
}

// ---------------- k4: fused softmax attention pooling (one block per batch) ----------------
__global__ __launch_bounds__(256) void k4_attn(
    const h16* __restrict__ hbuf,
    const float* __restrict__ Acoef, const float* __restrict__ Ccoef,
    const float* __restrict__ attn_w, const float* __restrict__ attn_b,
    float* __restrict__ out)
{
  __shared__ float sc[SS];        // 8 KB scores -> exp
  __shared__ float wA[HH];
  __shared__ float red[256];
  __shared__ float smax, sden, sc0;
  int b = blockIdx.x, t = threadIdx.x;

  float aw = attn_w[t];
  wA[t]  = Acoef[b * HH + t] * aw;
  red[t] = Ccoef[b * HH + t] * aw;
  __syncthreads();
  for (int off = 128; off > 0; off >>= 1) { if (t < off) red[t] += red[t + off]; __syncthreads(); }
  if (t == 0) sc0 = red[0] + attn_b[0];
  __syncthreads();
  float c0 = sc0;

  const h16* hb = hbuf + (size_t)b * SS * HH;
  float lmax = -3.0e38f, ls[8];
#pragma unroll
  for (int i = 0; i < 8; ++i) {
    int s = t + i * 256;
    const h16* row = hb + (size_t)s * HH;
    __builtin_prefetch(row + HH, 0, 1);          // global_prefetch of next row
    float a = c0;
    for (int h = 0; h < HH; ++h) a += (float)row[h] * wA[h];
    ls[i] = a;
    lmax = fmaxf(lmax, a);
  }
  red[t] = lmax; __syncthreads();
  for (int off = 128; off > 0; off >>= 1) { if (t < off) red[t] = fmaxf(red[t], red[t + off]); __syncthreads(); }
  if (t == 0) smax = red[0];
  __syncthreads();
  float mx = smax, lsum = 0.f;
#pragma unroll
  for (int i = 0; i < 8; ++i) {
    float e = __expf(ls[i] - mx);
    sc[t + i * 256] = e;
    lsum += e;
  }
  red[t] = lsum; __syncthreads();
  for (int off = 128; off > 0; off >>= 1) { if (t < off) red[t] += red[t + off]; __syncthreads(); }
  if (t == 0) sden = 1.0f / red[0];
  __syncthreads();
  float rd = sden;

  // channel-owned weighted sum; sum(prob)=1 keeps the affine fold exact
  float acc = 0.f;
  for (int s = 0; s < SS; ++s)
    acc += sc[s] * (float)hb[(size_t)s * HH + t];
  out[b * HH + t] = Acoef[b * HH + t] * acc * rd + Ccoef[b * HH + t];
}

// ---------------- launcher ----------------
extern "C" void kernel_launch(void* const* d_in, const int* in_sizes, int n_in,
                              void* d_out, int out_size, void* d_ws, size_t ws_size,
                              hipStream_t stream) {
  const float* raw_seq    = (const float*)d_in[0];
  const float* raw_stats  = (const float*)d_in[1];
  const float* mu         = (const float*)d_in[2];
  const float* log_sigma  = (const float*)d_in[3];
  const float* kernel_emb = (const float*)d_in[4];
  const float* sign_emb   = (const float*)d_in[5];
  const float* proj_w     = (const float*)d_in[6];
  const float* proj_b     = (const float*)d_in[7];
  const float* c3w = (const float*)d_in[8];
  const float* c3b = (const float*)d_in[9];
  const float* c5w = (const float*)d_in[10];
  const float* c5b = (const float*)d_in[11];
  const float* c7w = (const float*)d_in[12];
  const float* c7b = (const float*)d_in[13];
  const float* bn_gamma = (const float*)d_in[14];
  const float* bn_beta  = (const float*)d_in[15];
  const float* sw1 = (const float*)d_in[16];
  const float* sb1 = (const float*)d_in[17];
  const float* sw2 = (const float*)d_in[18];
  const float* sb2 = (const float*)d_in[19];
  const float* attn_w = (const float*)d_in[20];
  const float* attn_b = (const float*)d_in[21];
  float* out = (float*)d_out;

  size_t off = 0;
  auto carve = [&](size_t bytes) -> void* {
    void* p = (char*)d_ws + off;
    off += (bytes + 255) & ~(size_t)255;
    return p;
  };
  h16*   keQ     = (h16*)  carve((size_t)KK * EE * 2);
  h16*   projTQ  = (h16*)  carve((size_t)EE * EE * 2);
  h16*   W7Q     = (h16*)  carve((size_t)7 * EE * HH * 2);
  float* hbias   = (float*)carve((size_t)HH * 4);
  float* bnsum   = (float*)carve((size_t)HH * 4);
  float* bnsumsq = (float*)carve((size_t)HH * 4);
  float* Acoef   = (float*)carve((size_t)BB * HH * 4);
  float* Ccoef   = (float*)carve((size_t)BB * HH * 4);
  h16*   seqP    = (h16*)  carve((size_t)BB * SP * EE * 2);   // ~64 MB
  h16*   hbuf    = (h16*)  carve((size_t)BB * SS * HH * 2);   // 128 MB (L2-resident)

  k0_prep<<<512, 256, 0, stream>>>(kernel_emb, proj_w, c3w, c3b, c5w, c5b, c7w, c7b,
                                   keQ, projTQ, W7Q, hbias, bnsum, bnsumsq, seqP);
  k1_embed_proj<<<(BB * SS) / 128, 256, 0, stream>>>(raw_seq, mu, log_sigma, sign_emb, proj_b,
                                                     keQ, projTQ, seqP);
  k2_conv<<<(BB * SS) / 128, 128, 0, stream>>>(seqP, W7Q, hbias, hbuf, bnsum, bnsumsq);
  k3_film<<<BB, 256, 0, stream>>>(raw_stats, sw1, sb1, sw2, sb2, bn_gamma, bn_beta,
                                  bnsum, bnsumsq, Acoef, Ccoef);
  k4_attn<<<BB, 256, 0, stream>>>(hbuf, Acoef, Ccoef, attn_w, attn_b, out);
}
